// RocketConv1d_9148280340928
// MI455X (gfx1250) — compile-verified, matrix operand order checked by reading, not run on hardware
//
#include <hip/hip_runtime.h>
#include <hip/hip_bf16.h>

// ---------------------------------------------------------------------------
// RocketConv1d on gfx1250 via implicit-GEMM bf16 WMMA (v_wmma_f32_16x16x32_bf16)
//   A (16x32) = sliding windows of padded input signal (time on M axis)
//   B (32x16) = pre-swizzled weight fragments (kernel on N axis), reg-resident
//   C (16x16 f32) accumulated over 8 K-chunks (L=249 padded to 256)
// sched_barrier(0) pins all 16 A-loads of a tile ahead of the 8 WMMAs so the
// waits stagger (loadcnt<=14,12,...) and XDL overlaps outstanding loads.
// ---------------------------------------------------------------------------

typedef __attribute__((ext_vector_type(16))) __bf16        v16bf;
typedef __attribute__((ext_vector_type(8)))  float         v8f;
typedef __attribute__((ext_vector_type(8)))  unsigned int  v8u;
// 16-byte vector load with only 4-byte alignment guarantee (dword-mode VMEM ok)
typedef __attribute__((ext_vector_type(4), aligned(4))) unsigned int v4u_a4;

#define TT    2000      // time length
#define PAD   124       // (249-1)/2
#define LPAD  249       // dense kernel length
#define NCH   256       // B*C channels
#define NKT   16        // kernel tiles (256 kernels / 16)
#define XDW   1152      // dwords per packed copy per channel (covers 2304 elems)
#define NT    125       // time tiles (2000/16)
#define WB_BYTES (16*8*32*16*2)  // weight-fragment workspace: 131072 bytes

__device__ __forceinline__ unsigned int f2bf(float f) {
  unsigned int u = __float_as_uint(f);
  unsigned int r = u + 0x7FFFu + ((u >> 16) & 1u);   // round-to-nearest-even
  return r >> 16;
}

// --- Preprocess 1: pad x, convert to bf16, store two phase-packed copies ----
// xe[i] = pack(xp[2i],   xp[2i+1]) ; xo[i] = pack(xp[2i+1], xp[2i+2])
// where xp[j] = x[ch][j-PAD] inside bounds else 0.
__global__ void pack_x_kernel(const float* __restrict__ x,
                              unsigned int* __restrict__ xpk) {
  const int ch = blockIdx.x;
  const float* xc = x + ch * TT;
  unsigned int* xe = xpk + ch * (2 * XDW);
  unsigned int* xo = xe + XDW;
  for (int i = threadIdx.x; i < XDW; i += blockDim.x) {
    const int j0 = 2 * i;
    float a = (j0     >= PAD && j0     < PAD + TT) ? xc[j0     - PAD] : 0.0f;
    float b = (j0 + 1 >= PAD && j0 + 1 < PAD + TT) ? xc[j0 + 1 - PAD] : 0.0f;
    float c = (j0 + 2 >= PAD && j0 + 2 < PAD + TT) ? xc[j0 + 2 - PAD] : 0.0f;
    unsigned int ha = f2bf(a), hb = f2bf(b), hc = f2bf(c);
    xe[i] = ha | (hb << 16);
    xo[i] = hb | (hc << 16);
  }
}

// --- Preprocess 2: swizzle weights into 32x16 bf16 B-fragment layout --------
// Lane l (n=l&15, hi=l>>4) holds column n: K = {kc*32+8*hi+0..7, kc*32+16+8*hi+0..7}
// stored as 16 contiguous bf16 per lane (32B, aligned).
__global__ void pack_w_kernel(const float* __restrict__ w,
                              unsigned short* __restrict__ wb) {
  const int id   = blockIdx.x * blockDim.x + threadIdx.x;  // 65536 total
  const int e    = id & 15;
  const int lane = (id >> 4) & 31;
  const int kc   = (id >> 9) & 7;
  const int mt   = id >> 12;
  const int n  = lane & 15, hi = lane >> 4;
  const int kk = kc * 32 + ((e < 8) ? (8 * hi + e) : (16 + 8 * hi + (e - 8)));
  const float v = (kk < LPAD) ? w[(mt * 16 + n) * LPAD + kk] : 0.0f;
  wb[id] = (unsigned short)f2bf(v);
}

// --- Main kernel: block = (kernel-tile mt, channel ch), 4 waves -------------
__global__ __launch_bounds__(128) void rocket_wmma_kernel(
    const unsigned int*  __restrict__ xpk,
    const unsigned short* __restrict__ wb,
    const float* __restrict__ bias,
    float* __restrict__ out) {
  const int mt   = blockIdx.x;
  const int ch   = blockIdx.y;
  const int tid  = threadIdx.x;
  const int lane = tid & 31;
  const int wave = tid >> 5;
  const int n    = lane & 15;   // column-in-tile (kernel) / row-in-tile (time) id
  const int hi   = lane >> 4;

  // Weight B-fragments: register-resident for all 8 K-chunks (64 VGPRs)
  v16bf bfr[8];
#pragma unroll
  for (int kc = 0; kc < 8; ++kc) {
    union { v8u u; v16bf v; } cvt;
    cvt.u = *(const v8u*)(wb + ((mt * 8 + kc) * 32 + lane) * 16);  // 32B aligned
    bfr[kc] = cvt.v;
  }

  const float bn = bias[mt * 16 + n];

  // Per-lane base into the parity-matched packed copy (parity of m==n is fixed)
  const unsigned int* ptr = xpk + ch * (2 * XDW) + ((n & 1) ? XDW : 0);
  const int qbase = ((n - (n & 1)) >> 1) + 4 * hi;

  int   cnt  = 0;
  float vmax = -3.402823466e+38f;

  for (int nt = wave; nt < NT; nt += 4) {
    const int q0 = 8 * nt + qbase;

    // Stage ALL A-fragments for this tile: one 16-load clause.
    v4u_a4 ua[16];
#pragma unroll
    for (int kc = 0; kc < 8; ++kc) {
      const unsigned int* pp = ptr + q0 + 16 * kc;
      ua[2 * kc + 0] = *(const v4u_a4*)(pp + 0);   // elems o..o+7   (K lo half)
      ua[2 * kc + 1] = *(const v4u_a4*)(pp + 8);   // elems o+16..23 (K hi half)
    }

    // Hard scheduling fence: keep every load issued above this line so the
    // WMMAs below wait on staggered loadcnt instead of 0 each.
    __builtin_amdgcn_sched_barrier(0);

    v8f c = {0.f, 0.f, 0.f, 0.f, 0.f, 0.f, 0.f, 0.f};
#pragma unroll
    for (int kc = 0; kc < 8; ++kc) {
      union { struct { v4u_a4 lo, hi; } p; v16bf v; } a;
      a.p.lo = ua[2 * kc + 0];
      a.p.hi = ua[2 * kc + 1];
      c = __builtin_amdgcn_wmma_f32_16x16x32_bf16(
              /*neg_a=*/false, a.v, /*neg_b=*/false, bfr[kc],
              /*c_mod=*/(short)0, c, /*reuse_a=*/false, /*reuse_b=*/false);
    }

#pragma unroll
    for (int r = 0; r < 8; ++r) {
      const float y = c[r] + bn;
      cnt += (y > 0.0f) ? 1 : 0;
      vmax = fmaxf(vmax, y);
    }
  }

  // Combine the two M-halves (lane ^ 16 holds rows 8..15 of same kernel column)
  cnt  += __shfl_xor(cnt, 16, 32);
  vmax  = fmaxf(vmax, __shfl_xor(vmax, 16, 32));

  // Reduce the 4 waves through LDS; block owns 16 (ppv,max) outputs exclusively
  __shared__ int   scnt[4][16];
  __shared__ float smax[4][16];
  if (lane < 16) { scnt[wave][lane] = cnt; smax[wave][lane] = vmax; }
  __syncthreads();
  if (tid < 16) {
    const int   c0 = scnt[0][tid] + scnt[1][tid] + scnt[2][tid] + scnt[3][tid];
    const float m0 = fmaxf(fmaxf(smax[0][tid], smax[1][tid]),
                           fmaxf(smax[2][tid], smax[3][tid]));
    const int k = mt * 16 + tid;
    out[ch * 512 + 2 * k + 0] = (float)c0 * (1.0f / 2000.0f);
    out[ch * 512 + 2 * k + 1] = m0;
  }
}

extern "C" void kernel_launch(void* const* d_in, const int* in_sizes, int n_in,
                              void* d_out, int out_size, void* d_ws, size_t ws_size,
                              hipStream_t stream) {
  (void)in_sizes; (void)n_in; (void)out_size; (void)ws_size;
  const float* x    = (const float*)d_in[0];   // (8,32,2000) f32
  const float* w    = (const float*)d_in[1];   // (256,249)  f32
  const float* bias = (const float*)d_in[2];   // (256,)     f32
  float* out = (float*)d_out;                  // (8,32,512) f32

  unsigned short* wb = (unsigned short*)d_ws;                       // 128 KB
  unsigned int*  xpk = (unsigned int*)((char*)d_ws + WB_BYTES);     // 2.25 MB

  pack_x_kernel<<<NCH, 256, 0, stream>>>(x, xpk);
  pack_w_kernel<<<256, 256, 0, stream>>>(w, wb);
  rocket_wmma_kernel<<<dim3(NKT, NCH), 128, 0, stream>>>(xpk, wb, bias, out);
}
